// ramsey_MPNN_2911987826887
// MI455X (gfx1250) — matrix operand order repaired and checked by reading, not compiled, not run on hardware
//
#include <hip/hip_runtime.h>

typedef _Float16 half16 __attribute__((ext_vector_type(16)));
typedef float    float8 __attribute__((ext_vector_type(8)));

union HU { uint4 q; _Float16 h[8]; };

#define LRELU_SLOPE 0.01f

// ---------------- Phase A: node MLP, fp32 exact ----------------

// out[M,Nn] = leaky_relu(A[M,K] @ W[K,Nn] + b)
__global__ void k_lin_lrelu(const float* __restrict__ A, const float* __restrict__ W,
                            const float* __restrict__ b, float* __restrict__ out,
                            int M, int K, int Nn) {
  int idx = blockIdx.x * blockDim.x + threadIdx.x;
  if (idx >= M * Nn) return;
  int row = idx / Nn, col = idx % Nn;
  const float* a = A + row * K;
  float s = 0.f;
  for (int k = 0; k < K; ++k) s = fmaf(a[k], W[k * Nn + col], s);
  s += b[col];
  out[idx] = s > 0.f ? s : LRELU_SLOPE * s;
}

// h16[1024,128] = (f16)( h2[1024,512] @ w4[512,128] + b4 + x0 )
__global__ void k_lin_res_h16(const float* __restrict__ h2, const float* __restrict__ w4,
                              const float* __restrict__ b4, const float* __restrict__ x0,
                              _Float16* __restrict__ h16) {
  int idx = blockIdx.x * blockDim.x + threadIdx.x;  // 1024*128
  int row = idx >> 7, col = idx & 127;
  const float* a = h2 + row * 512;
  float s = 0.f;
  for (int k = 0; k < 512; ++k) s = fmaf(a[k], w4[k * 128 + col], s);
  s += b4[col] + x0[idx];
  h16[idx] = (_Float16)s;
}

// w5t[512,128] f16 = transpose of w5[128,512] f32
__global__ void k_w5t(const float* __restrict__ w5, _Float16* __restrict__ w5t) {
  int idx = blockIdx.x * blockDim.x + threadIdx.x;  // 128*512, coalesced reads
  int k = idx >> 9, n = idx & 511;
  w5t[n * 128 + k] = (_Float16)w5[idx];
}

__global__ void k_zero_diag(float* __restrict__ out) {
  int idx = blockIdx.x * blockDim.x + threadIdx.x;  // 2048
  int i = idx >> 1, c = idx & 1;
  out[(i * 1024 + i) * 2 + c] = 0.f;
}

// ---------------- Phase B: edge MLP via WMMA f16->f32 ----------------

__global__ void __launch_bounds__(256)
edge_kernel(const _Float16* __restrict__ h16, const _Float16* __restrict__ w5t,
            const float* __restrict__ b5, const float* __restrict__ w6,
            const float* __restrict__ b6, float* __restrict__ out) {
  const int I = blockIdx.x, J = blockIdx.y;
  if (J < I) return;

  extern __shared__ char smem[];
  _Float16* sw5 = (_Float16*)smem;               // [512*128] halves = 128 KB
  float*    sb5 = (float*)(smem + 65536 * 2);    // [512]
  float*    sw6 = sb5 + 512;                     // [512*2]

  {  // cooperative stage of w5t / b5 / w6 into LDS
    const uint4* src = (const uint4*)w5t;
    uint4* dst = (uint4*)sw5;
    for (int i = threadIdx.x; i < 8192; i += 256) dst[i] = src[i];
    for (int i = threadIdx.x; i < 512; i += 256) sb5[i] = b5[i];
    for (int i = threadIdx.x; i < 1024; i += 256) sw6[i] = w6[i];
  }
  __syncthreads();

  const int wave = threadIdx.x >> 5;
  const int lane = threadIdx.x & 31;
  const int m    = lane & 15;
  const bool hih = lane >= 16;

  // two edge tiles per wave: i0 = 16I+wave, i1 = 16I+wave+8 ; rows j = 16J+m
  const int i0 = I * 16 + wave;
  const int i1 = I * 16 + wave + 8;
  const int jm = J * 16 + m;

  // Build A fragments (contiguous k order, same as B: lanes<16 K={0..15}, lanes>=16 K={16..31})
  half16 A0[4], A1[4];
  const int aoff = hih ? 16 : 0;
#pragma unroll
  for (int ks = 0; ks < 4; ++ks) {
    const int kb = ks * 32 + aoff;
    HU hj0, hj1, ha, hb;
    hj0.q = *(const uint4*)(h16 + jm * 128 + kb);
    hj1.q = *(const uint4*)(h16 + jm * 128 + kb + 8);
    ha.q  = *(const uint4*)(h16 + i0 * 128 + kb);
    hb.q  = *(const uint4*)(h16 + i0 * 128 + kb + 8);
#pragma unroll
    for (int e = 0; e < 8; ++e) {
      A0[ks][e]     = ha.h[e] * hj0.h[e];
      A0[ks][e + 8] = hb.h[e] * hj1.h[e];
    }
    ha.q = *(const uint4*)(h16 + i1 * 128 + kb);
    hb.q = *(const uint4*)(h16 + i1 * 128 + kb + 8);
#pragma unroll
    for (int e = 0; e < 8; ++e) {
      A1[ks][e]     = ha.h[e] * hj0.h[e];
      A1[ks][e + 8] = hb.h[e] * hj1.h[e];
    }
  }

  // fused second layer accumulators: logits per row (8 rows per lane-half), 2 classes, 2 tiles
  float l00[8], l01[8], l10[8], l11[8];
#pragma unroll
  for (int v = 0; v < 8; ++v) { l00[v] = 0.f; l01[v] = 0.f; l10[v] = 0.f; l11[v] = 0.f; }

  const int boff = hih ? 16 : 0;
  for (int nt = 0; nt < 32; ++nt) {
    const int n_g = nt * 16 + m;
    const _Float16* bp = sw5 + n_g * 128 + boff;

    // hoist ALL B fragments first: 8 back-to-back ds_load_b128, single wait,
    // then an uninterrupted burst of 8 WMMAs (two independent acc chains)
    half16 Bv[4];
#pragma unroll
    for (int ks = 0; ks < 4; ++ks) Bv[ks] = *(const half16*)(bp + ks * 32);

    float8 acc0 = {}, acc1 = {};
#pragma unroll
    for (int ks = 0; ks < 4; ++ks) {
      acc0 = __builtin_amdgcn_wmma_f32_16x16x32_f16(false, A0[ks], false, Bv[ks],
                                                    (short)0, acc0, false, false);
      acc1 = __builtin_amdgcn_wmma_f32_16x16x32_f16(false, A1[ks], false, Bv[ks],
                                                    (short)0, acc1, false, false);
    }

    const float bias = sb5[n_g];
    const float w60 = sw6[2 * n_g], w61 = sw6[2 * n_g + 1];
#pragma unroll
    for (int v = 0; v < 8; ++v) {
      float r0 = fmaxf(acc0[v] + bias, 0.f);
      l00[v] = fmaf(r0, w60, l00[v]);
      l01[v] = fmaf(r0, w61, l01[v]);
      float r1 = fmaxf(acc1[v] + bias, 0.f);
      l10[v] = fmaf(r1, w60, l10[v]);
      l11[v] = fmaf(r1, w61, l11[v]);
    }
  }

  // butterfly reduce over the 16 lanes of each half (each lane held one n-column residue)
#pragma unroll
  for (int off = 1; off < 16; off <<= 1) {
#pragma unroll
    for (int v = 0; v < 8; ++v) {
      l00[v] += __shfl_xor(l00[v], off, 32);
      l01[v] += __shfl_xor(l01[v], off, 32);
      l10[v] += __shfl_xor(l10[v], off, 32);
      l11[v] += __shfl_xor(l11[v], off, 32);
    }
  }

  const float b60 = b6[0], b61 = b6[1];
  const int c = m;  // class index carried by lane
  if (c < 2) {
#pragma unroll
    for (int v = 0; v < 8; ++v) {
      const int mr = v + (hih ? 8 : 0);
      const int jn = J * 16 + mr;
      {  // tile 0
        float z0 = l00[v] + b60, z1 = l01[v] + b61;
        float mx = fmaxf(z0, z1);
        float e0 = __expf(z0 - mx), e1 = __expf(z1 - mx);
        float p = (c == 0 ? e0 : e1) / (e0 + e1);
        if (i0 < jn) {
          out[(i0 * 1024 + jn) * 2 + c] = p;
          out[(jn * 1024 + i0) * 2 + c] = p;
        }
      }
      {  // tile 1
        float z0 = l10[v] + b60, z1 = l11[v] + b61;
        float mx = fmaxf(z0, z1);
        float e0 = __expf(z0 - mx), e1 = __expf(z1 - mx);
        float p = (c == 0 ? e0 : e1) / (e0 + e1);
        if (i1 < jn) {
          out[(i1 * 1024 + jn) * 2 + c] = p;
          out[(jn * 1024 + i1) * 2 + c] = p;
        }
      }
    }
  }
}

// ---------------- Host launch ----------------

extern "C" void kernel_launch(void* const* d_in, const int* in_sizes, int n_in,
                              void* d_out, int out_size, void* d_ws, size_t ws_size,
                              hipStream_t stream) {
  const float* x  = (const float*)d_in[1];   // node_features [1024,128]
  const float* w1 = (const float*)d_in[2];   // [128,512]
  const float* b1 = (const float*)d_in[3];
  const float* w2 = (const float*)d_in[4];   // [512,512]
  const float* b2 = (const float*)d_in[5];
  const float* w4 = (const float*)d_in[6];   // [512,128]
  const float* b4 = (const float*)d_in[7];
  const float* w5 = (const float*)d_in[8];   // [128,512]
  const float* b5 = (const float*)d_in[9];
  const float* w6 = (const float*)d_in[10];  // [512,2]
  const float* b6 = (const float*)d_in[11];
  float* out = (float*)d_out;                // [1024,1024,2]

  char* ws = (char*)d_ws;
  float*    h1  = (float*)ws;                              // 2 MB
  float*    h2  = (float*)(ws + (1u << 21));               // 2 MB
  _Float16* h16 = (_Float16*)(ws + (1u << 22));            // 256 KB
  _Float16* w5t = (_Float16*)(ws + (1u << 22) + (1u << 18));  // 128 KB

  k_lin_lrelu<<<2048, 256, 0, stream>>>(x,  w1, b1, h1, 1024, 128, 512);
  k_lin_lrelu<<<2048, 256, 0, stream>>>(h1, w2, b2, h2, 1024, 512, 512);
  k_lin_res_h16<<<512, 256, 0, stream>>>(h2, w4, b4, x, h16);
  k_w5t<<<256, 256, 0, stream>>>(w5, w5t);
  k_zero_diag<<<8, 256, 0, stream>>>(out);

  dim3 grid(64, 64);
  size_t shmem = 65536 * 2 + 512 * 4 + 1024 * 4;  // 137216 B (<320 KB WGP LDS)
  edge_kernel<<<grid, 256, shmem, stream>>>(h16, w5t, b5, w6, b6, out);
}
